// LearnableHash_23845658428386
// MI455X (gfx1250) — compile-verified, matrix operand order checked by reading, not run on hardware
//
#include <hip/hip_runtime.h>
#include <hip/hip_bf16.h>

typedef __attribute__((ext_vector_type(16))) _Float16 v16h;
typedef __attribute__((ext_vector_type(8)))  float    v8f;

#define PTS_PER_BLOCK 128
#define GRES 256
#define NFEAT 65536

// K index for element e of a v16h A/B fragment held by `lane`
// (CDNA5 16-bit 16x32 A layout; B mirrors it with N in place of M).
__device__ __forceinline__ int kmap(int lane, int e) {
  int hi = (lane & 16) ? 8 : 0;
  return (e < 8) ? (hi + e) : (16 + hi + (e - 8));
}

// Build one pre-swizzled B fragment (32x16 K x N tile) for simulated lane `sl`.
// Fragment table:
//   0..3  : Ws1 (16x64, K-pad->32), nOff = f*16
//   4..5  : Ws2 (64x16), kOff = (f-4)*32
//   6..9  : Wc1 (31x64, K-pad->32), nOff = (f-6)*16
//  10..17 : Wc2 (64x64), kOff = ((f-10)&1)*32, nOff = ((f-10)>>1)*16
//  18..19 : Wc3 (64x3, N-pad->16), kOff = (f-18)*32
__device__ __forceinline__ v16h make_Bfrag(int f, int sl, const float* Ws1,
                                           const float* Ws2, const float* Wc1,
                                           const float* Wc2, const float* Wc3) {
  const float* W;
  int stride, Krows, Ncols, kOff, nOff;
  if (f < 4)       { W = Ws1; stride = 64; Krows = 16; Ncols = 64; kOff = 0;               nOff = f * 16; }
  else if (f < 6)  { W = Ws2; stride = 16; Krows = 64; Ncols = 16; kOff = (f - 4) * 32;    nOff = 0; }
  else if (f < 10) { W = Wc1; stride = 64; Krows = 31; Ncols = 64; kOff = 0;               nOff = (f - 6) * 16; }
  else if (f < 18) { W = Wc2; stride = 64; Krows = 64; Ncols = 64; kOff = ((f - 10) & 1) * 32; nOff = ((f - 10) >> 1) * 16; }
  else             { W = Wc3; stride = 3;  Krows = 64; Ncols = 3;  kOff = (f - 18) * 32;   nOff = 0; }
  int n = nOff + (sl & 15);
  v16h b;
#pragma unroll
  for (int e = 0; e < 16; ++e) {
    int k = kmap(sl, e) + kOff;
    float v = (k < Krows && n < Ncols) ? W[k * stride + n] : 0.f;
    b[e] = (_Float16)v;
  }
  return b;
}

// Scatter a 16x16 f32 C fragment (cols nOff..nOff+15 of a M x 64 activation)
// into an A-fragment-layout LDS buffer scrA = [2 kfrags][32 lanes][16 halves].
// col is fixed per lane, so kfrag/e/lane-half are per-lane constants.
__device__ __forceinline__ void store_C_scrA(_Float16* scrA, int nOff, int lane,
                                             v8f c, bool relu) {
  int col = nOff + (lane & 15);
  int rb = (lane & 16) ? 8 : 0;
  int kk = col >> 5;
  int krel = col & 31;
  int halfAdd = (krel & 8) ? 16 : 0;                 // which lane-half holds this k
  int e = (krel & 7) + ((krel & 16) ? 8 : 0);        // element slot within v16h
  _Float16* base = scrA + kk * (32 * 16) + halfAdd * 16 + e;
#pragma unroll
  for (int i = 0; i < 8; ++i) {
    float v = c[i];
    if (relu) v = fmaxf(v, 0.f);
    base[(rb + i) * 16] = (_Float16)v;
  }
}

__global__ __launch_bounds__(PTS_PER_BLOCK) void lh_fused_kernel(
    const float* __restrict__ pts, const float* __restrict__ rds,
    const float* __restrict__ G, const float* __restrict__ F,
    const float* __restrict__ Ws1, const float* __restrict__ Ws2,
    const float* __restrict__ Wc1, const float* __restrict__ Wc2,
    const float* __restrict__ Wc3, float* __restrict__ out, int N) {
  __shared__ v16h sB[20][32];     // pre-swizzled weight B fragments (20 KB)
  __shared__ v16h sA[8][32];      // Fvals A fragments, one per 16-pt tile (8 KB)
  __shared__ v16h sEncA[8][32];   // c_in fragments: SH enc (e0..7) + Fvals2 (e8..15)
  __shared__ v16h sScr[4][2][32]; // per-wave intermediate activations, A layout

  const int tid = threadIdx.x;
  const int lane = tid & 31;
  const int wid = tid >> 5;

  // ---- stage weights: global f32 -> LDS f16, directly in B-fragment layout ----
  for (int idx = tid; idx < 20 * 32; idx += PTS_PER_BLOCK) {
    int f = idx >> 5, sl = idx & 31;
    sB[f][sl] = make_Bfrag(f, sl, Ws1, Ws2, Wc1, Wc2, Wc3);
  }

  // ---------------- phase 1: per-point gather / featurize ----------------
  int p = blockIdx.x * PTS_PER_BLOCK + tid;
  float fv[16], en[16];
#pragma unroll
  for (int k = 0; k < 16; ++k) { fv[k] = 0.f; en[k] = 0.f; }
  if (p < N) {
    float px = pts[3 * p + 0], py = pts[3 * p + 1], pz = pts[3 * p + 2];
    float ox = px - fminf(fmaxf(floorf(px - 0.5f), 0.f), (float)(GRES - 1));
    float oy = py - fminf(fmaxf(floorf(py - 0.5f), 0.f), (float)(GRES - 1));
    float oz = pz - fminf(fmaxf(floorf(pz - 0.5f), 0.f), (float)(GRES - 1));
#pragma unroll
    for (int j = 0; j < 8; ++j) {
      float sx = (j & 4) ? 0.5f : -0.5f;
      float sy = (j & 2) ? 0.5f : -0.5f;
      float sz = (j & 1) ? 0.5f : -0.5f;
      int ix = (int)fminf(fmaxf(floorf(px + sx), 0.f), (float)(GRES - 1));
      int iy = (int)fminf(fmaxf(floorf(py + sy), 0.f), (float)(GRES - 1));
      int iz = (int)fminf(fmaxf(floorf(pz + sz), 0.f), (float)(GRES - 1));
      float g = G[(ix << 16) + (iy << 8) + iz];
      g = fminf(fmaxf(g, -1.f), 1.f);
      float Fi = (g + 1.f) * ((float)NFEAT * 0.5f);
      float Fif = fminf(fmaxf(floorf(Fi), 0.f), (float)(NFEAT - 1));
      float Fic = fminf(fmaxf(ceilf(Fi), 0.f), (float)(NFEAT - 1));
      float wlo = Fic - Fi, whi = Fi - Fif;
      float wj = ((j & 4) ? ox : 1.f - ox) * ((j & 2) ? oy : 1.f - oy) *
                 ((j & 1) ? oz : 1.f - oz);
      const float4* rlo = (const float4*)(F + (((long)(int)Fif) << 4));
      const float4* rhi = (const float4*)(F + (((long)(int)Fic) << 4));
#pragma unroll
      for (int q = 0; q < 4; ++q) {
        float4 a = rlo[q], b = rhi[q];
        fv[4 * q + 0] += wj * (a.x * wlo + b.x * whi);
        fv[4 * q + 1] += wj * (a.y * wlo + b.y * whi);
        fv[4 * q + 2] += wj * (a.z * wlo + b.z * whi);
        fv[4 * q + 3] += wj * (a.w * wlo + b.w * whi);
      }
    }
    float x = rds[3 * p + 0], y = rds[3 * p + 1], z = rds[3 * p + 2];
    float x2 = x * x, y2 = y * y, z2 = z * z;
    en[0] = 0.28209479177387814f;
    en[1] = -0.48860251190291987f * y;
    en[2] = 0.48860251190291987f * z;
    en[3] = -0.48860251190291987f * x;
    en[4] = 1.0925484305920792f * x * y;
    en[5] = -1.0925484305920792f * y * z;
    en[6] = 0.94617469575756f * z2 - 0.31539156525252005f;
    en[7] = -1.0925484305920792f * x * z;
    en[8] = 0.5462742152960396f * (x2 - y2);
    en[9] = 0.5900435899266435f * y * (-3.f * x2 + y2);
    en[10] = 2.890611442640554f * x * y * z;
    en[11] = 0.4570457994644657f * y * (1.f - 5.f * z2);
    en[12] = 0.3731763325901154f * z * (5.f * z2 - 3.f);
    en[13] = 0.4570457994644657f * x * (1.f - 5.f * z2);
    en[14] = 1.445305721320277f * z * (x2 - y2);
    en[15] = 0.5900435899266435f * x * (-x2 + 3.f * y2);
  }
  // pack into A-fragment layout: lane m holds k=0..7, lane m+16 holds k=8..15
  {
    int tile = tid >> 4, m = tid & 15;
    v16h lo, hi, elo, ehi;
#pragma unroll
    for (int e = 0; e < 8; ++e) {
      lo[e] = (_Float16)fv[e];       hi[e] = (_Float16)fv[8 + e];
      elo[e] = (_Float16)en[e];      ehi[e] = (_Float16)en[8 + e];
      lo[8 + e] = (_Float16)0;       hi[8 + e] = (_Float16)0;
      elo[8 + e] = (_Float16)0;      ehi[8 + e] = (_Float16)0;   // k=16..31 pad / fv2 slot
    }
    sA[tile][m] = lo;      sA[tile][m + 16] = hi;
    sEncA[tile][m] = elo;  sEncA[tile][m + 16] = ehi;
  }
  __syncthreads();

  // ---------------- phase 2: WMMA MLP, two 16-point tiles per wave ----------------
  _Float16* scrA = (_Float16*)&sScr[wid][0][0];
  for (int t = wid * 2; t < wid * 2 + 2; ++t) {
    const int tb = blockIdx.x * PTS_PER_BLOCK + t * 16;

    // layer 1: h = relu(Fvals @ Ws1)   (16 -> 64)
    v16h a0 = sA[t][lane];
#pragma unroll
    for (int n = 0; n < 4; ++n) {
      v8f c = {};
      c = __builtin_amdgcn_wmma_f32_16x16x32_f16(false, a0, false, sB[n][lane],
                                                 (short)0, c, false, false);
      store_C_scrA(scrA, n * 16, lane, c, true);
    }

    // layer 2: Fvals2 = h @ Ws2   (64 -> 16)
    v16h h0 = sScr[wid][0][lane];
    v16h h1 = sScr[wid][1][lane];
    {
      v8f c = {};
      c = __builtin_amdgcn_wmma_f32_16x16x32_f16(false, h0, false, sB[4][lane],
                                                 (short)0, c, false, false);
      c = __builtin_amdgcn_wmma_f32_16x16x32_f16(false, h1, false, sB[5][lane],
                                                 (short)0, c, false, false);
      int cidx = lane & 15;
      int rb = (lane & 16) ? 8 : 0;
      if (cidx == 0) {  // sigma = Fvals2[:,0] -> global
#pragma unroll
        for (int i = 0; i < 8; ++i) {
          int pp = tb + rb + i;
          if (pp < N) out[pp] = c[i];
        }
      } else {  // Fvals2[:,1:16] -> c_in fragment slots (e = 8..15)
        int halfAdd = (cidx >= 9) ? 16 : 0;
        int e = (cidx >= 9) ? (8 + cidx - 9) : (8 + cidx - 1);
        _Float16* base = (_Float16*)&sEncA[t][0] + halfAdd * 16 + e;
#pragma unroll
        for (int i = 0; i < 8; ++i) base[(rb + i) * 16] = (_Float16)c[i];
      }
    }

    // layer 3: c1 = relu(c_in @ Wc1)   (31 -> 64)
    v16h ac = sEncA[t][lane];
#pragma unroll
    for (int n = 0; n < 4; ++n) {
      v8f c = {};
      c = __builtin_amdgcn_wmma_f32_16x16x32_f16(false, ac, false, sB[6 + n][lane],
                                                 (short)0, c, false, false);
      store_C_scrA(scrA, n * 16, lane, c, true);
    }

    // layer 4: c2 = relu(c1 @ Wc2)   (64 -> 64)
    v16h c1a = sScr[wid][0][lane];
    v16h c1b = sScr[wid][1][lane];
#pragma unroll
    for (int n = 0; n < 4; ++n) {
      v8f c = {};
      c = __builtin_amdgcn_wmma_f32_16x16x32_f16(false, c1a, false, sB[10 + n * 2][lane],
                                                 (short)0, c, false, false);
      c = __builtin_amdgcn_wmma_f32_16x16x32_f16(false, c1b, false, sB[11 + n * 2][lane],
                                                 (short)0, c, false, false);
      store_C_scrA(scrA, n * 16, lane, c, true);
    }

    // layer 5: colors = c2 @ Wc3   (64 -> 3)
    v16h c2a = sScr[wid][0][lane];
    v16h c2b = sScr[wid][1][lane];
    {
      v8f c = {};
      c = __builtin_amdgcn_wmma_f32_16x16x32_f16(false, c2a, false, sB[18][lane],
                                                 (short)0, c, false, false);
      c = __builtin_amdgcn_wmma_f32_16x16x32_f16(false, c2b, false, sB[19][lane],
                                                 (short)0, c, false, false);
      int n = lane & 15;
      if (n < 3) {
        int rb = (lane & 16) ? 8 : 0;
#pragma unroll
        for (int i = 0; i < 8; ++i) {
          int pp = tb + rb + i;
          if (pp < N) out[N + pp * 3 + n] = c[i];
        }
      }
    }
  }
}

extern "C" void kernel_launch(void* const* d_in, const int* in_sizes, int n_in,
                              void* d_out, int out_size, void* d_ws, size_t ws_size,
                              hipStream_t stream) {
  (void)n_in; (void)out_size; (void)d_ws; (void)ws_size;
  const float* pts = (const float*)d_in[0];
  const float* rds = (const float*)d_in[1];
  const float* G   = (const float*)d_in[2];
  const float* F   = (const float*)d_in[3];
  const float* Ws1 = (const float*)d_in[4];
  const float* Ws2 = (const float*)d_in[5];
  const float* Wc1 = (const float*)d_in[6];
  const float* Wc2 = (const float*)d_in[7];
  const float* Wc3 = (const float*)d_in[8];
  float* out = (float*)d_out;
  const int N = in_sizes[0] / 3;
  const int blocks = (N + PTS_PER_BLOCK - 1) / PTS_PER_BLOCK;
  lh_fused_kernel<<<blocks, PTS_PER_BLOCK, 0, stream>>>(
      pts, rds, G, F, Ws1, Ws2, Wc1, Wc2, Wc3, out, N);
}
